// PointNet2SAModuleCUDA_5007931867447
// MI455X (gfx1250) — matrix operand order repaired and checked by reading, not compiled, not run on hardware
//
#include <hip/hip_runtime.h>

// ---------------- problem constants ----------------
#define B_    4
#define N_    8192
#define NP_   2048
#define NS_   32
#define CIN_  64
#define R2_   0.01f          // RADIUS^2
#define CNT_  (B_*NP_*NS_)   // 262144 rows through the MLP
#define EPS_  1e-5f

typedef _Float16 half_t;
typedef _Float16 v16h __attribute__((ext_vector_type(16)));
typedef float    v8f  __attribute__((ext_vector_type(8)));

// =====================================================================
// K0: zero the BN-stats accumulators (must run every launch)
// =====================================================================
__global__ void pn2_zero_stats(float* stats) {
    if (threadIdx.x < 512) stats[threadIdx.x] = 0.f;
}

// =====================================================================
// K1: farthest-point sampling. One block per batch, points register-
// resident (1024 thr x 8 pts). Also writes new_xyz (output part 1).
// =====================================================================
__global__ __launch_bounds__(1024) void pn2_fps(
    const float* __restrict__ xyz, int* __restrict__ fpsIdx,
    float* __restrict__ newxyz)
{
    __shared__ float rD[32];
    __shared__ int   rI[32];
    __shared__ float curP[3];
    __shared__ int   curIdx;

    const int b   = blockIdx.x;
    const int tid = threadIdx.x;
    const float* base = xyz + (size_t)b * N_ * 3;

    float px[8], py[8], pz[8], dist[8];
#pragma unroll
    for (int j = 0; j < 8; ++j) {
        int p = tid + j * 1024;
        px[j] = base[p*3+0]; py[j] = base[p*3+1]; pz[j] = base[p*3+2];
        dist[j] = 1e10f;
    }
    if (tid == 0) {
        fpsIdx[b*NP_] = 0;
        curP[0] = base[0]; curP[1] = base[1]; curP[2] = base[2];
        newxyz[(size_t)(b*NP_)*3+0] = base[0];
        newxyz[(size_t)(b*NP_)*3+1] = base[1];
        newxyz[(size_t)(b*NP_)*3+2] = base[2];
    }
    __syncthreads();

    const int lane = tid & 31, wv = tid >> 5;
    for (int it = 1; it < NP_; ++it) {
        float cx = curP[0], cy = curP[1], cz = curP[2];
        float bd = -1.f; int bi = 0x7fffffff;
#pragma unroll
        for (int j = 0; j < 8; ++j) {
            float dx = px[j]-cx, dy = py[j]-cy, dz = pz[j]-cz;
            float d = dx*dx + dy*dy + dz*dz;
            dist[j] = fminf(dist[j], d);
            if (dist[j] > bd) { bd = dist[j]; bi = tid + j*1024; }
        }
        // wave32 argmax, prefer smaller index on ties (matches jnp.argmax)
        for (int off = 16; off > 0; off >>= 1) {
            float od = __shfl_down(bd, off, 32);
            int   oi = __shfl_down(bi, off, 32);
            if (od > bd || (od == bd && oi < bi)) { bd = od; bi = oi; }
        }
        if (lane == 0) { rD[wv] = bd; rI[wv] = bi; }
        __syncthreads();
        if (wv == 0) {
            bd = rD[lane]; bi = rI[lane];
            for (int off = 16; off > 0; off >>= 1) {
                float od = __shfl_down(bd, off, 32);
                int   oi = __shfl_down(bi, off, 32);
                if (od > bd || (od == bd && oi < bi)) { bd = od; bi = oi; }
            }
            if (lane == 0) { curIdx = bi; fpsIdx[b*NP_ + it] = bi; }
        }
        __syncthreads();
        int ci = curIdx;
        if (tid == (ci & 1023)) {         // owner broadcasts coords from regs
            int j = ci >> 10;
            curP[0] = px[j]; curP[1] = py[j]; curP[2] = pz[j];
            newxyz[(size_t)(b*NP_+it)*3+0] = px[j];
            newxyz[(size_t)(b*NP_+it)*3+1] = py[j];
            newxyz[(size_t)(b*NP_+it)*3+2] = pz[j];
        }
        __syncthreads();
    }
}

// =====================================================================
// K2: ball query. One wave per center; ordered stream compaction via
// ballot -> first 32 indices (ascending) inside radius, fill with first.
// =====================================================================
__global__ __launch_bounds__(256) void pn2_ball(
    const float* __restrict__ xyz, const int* __restrict__ fpsIdx,
    int* __restrict__ ballIdx)
{
    const int gw   = blockIdx.x * 8 + (threadIdx.x >> 5);   // center id
    const int lane = threadIdx.x & 31;
    const int b = gw / NP_;
    const float* base = xyz + (size_t)b * N_ * 3;
    const int ctr = fpsIdx[gw];
    const float cx = base[ctr*3+0], cy = base[ctr*3+1], cz = base[ctr*3+2];

    int cnt = 0, firstIdx = 0;
    int* out = ballIdx + (size_t)gw * NS_;
    for (int s0 = 0; s0 < N_ && cnt < NS_; s0 += 32) {
        int pi = s0 + lane;
        float dx = base[pi*3+0]-cx, dy = base[pi*3+1]-cy, dz = base[pi*3+2]-cz;
        bool in = (dx*dx + dy*dy + dz*dz) < R2_;
        unsigned mask = (unsigned)__ballot(in);
        if (cnt == 0 && mask) firstIdx = s0 + __builtin_ctz(mask);
        if (in) {
            int pos = cnt + __popc(mask & ((1u << lane) - 1u));
            if (pos < NS_) out[pos] = pi;
        }
        cnt += __popc(mask);
    }
    if (cnt < NS_ && lane >= cnt) out[lane] = firstIdx;  // firstIdx==0 if none
}

// -------- WMMA fragment loaders (ISA §7.12.2 layouts) ----------------
__device__ __forceinline__ v16h frag_a(const half_t* tile, int mrow,
                                       int stride, int kk, int lane) {
    const half_t* p = tile + (mrow + (lane & 15)) * stride + kk + 8*(lane >> 4);
    v16h f;
#pragma unroll
    for (int i = 0; i < 8; ++i) { f[i] = p[i]; f[8+i] = p[16+i]; }
    return f;
}
__device__ __forceinline__ v16h frag_b(const half_t* wtile, int ncol,
                                       int stride, int kk, int lane) {
    const half_t* p = wtile + (ncol + (lane & 15)) * stride + kk + 8*(lane >> 4);
    v16h f;
#pragma unroll
    for (int i = 0; i < 8; ++i) { f[i] = p[i]; f[8+i] = p[16+i]; }
    return f;
}

// =====================================================================
// K3: layer0 = gather(g_xyz|g_feat) -> WMMA GEMM (K=67 pad 96, CO=64)
// + bias, emit h0 (f16) + per-channel sum/sumsq partials.
// =====================================================================
__global__ __launch_bounds__(256) void pn2_gemm0(
    const float* __restrict__ xyz, const float* __restrict__ feat,
    const float* __restrict__ newxyz, const int* __restrict__ ballIdx,
    const float* __restrict__ W0, const float* __restrict__ b0,
    half_t* __restrict__ h0, float* __restrict__ sum0, float* __restrict__ sq0)
{
    __shared__ half_t Ah[128*96];
    __shared__ half_t Wh[64*96];
    __shared__ float  sS[8][64];
    __shared__ float  sQ[8][64];

    const int tid  = threadIdx.x;
    const int row0 = blockIdx.x * 128;

    for (int i = tid; i < 64*96; i += 256) {            // stage W0 (pad K)
        int o = i / 96, c = i % 96;
        Wh[i] = (half_t)((c < 67) ? W0[o*67 + c] : 0.f);
    }
    {                                                   // build A tile
        int t = tid >> 1, hsel = tid & 1;
        int r = row0 + t;
        int b = r / (NP_*NS_);
        int rem = r % (NP_*NS_);
        int p = rem / NS_, s = rem % NS_;
        (void)s;
        int id = ballIdx[r];
        const float* nx = newxyz + (size_t)(b*NP_ + p) * 3;
        const float* px = xyz + (size_t)(b*N_ + id) * 3;
        for (int c = hsel*48; c < hsel*48 + 48; ++c) {
            float v;
            if (c < 3)       v = px[c] - nx[c];
            else if (c < 67) v = feat[((size_t)b*CIN_ + (c-3)) * N_ + id];
            else             v = 0.f;
            Ah[t*96 + c] = (half_t)v;
        }
    }
    __syncthreads();

    const int wave = tid >> 5, lane = tid & 31;
    const int lrow = lane & 15, lhalf = lane >> 4;

    v8f acc[4];
#pragma unroll
    for (int t = 0; t < 4; ++t)
#pragma unroll
        for (int j = 0; j < 8; ++j) acc[t][j] = 0.f;

#pragma unroll
    for (int kk = 0; kk < 96; kk += 32) {
        v16h a = frag_a(Ah, wave*16, 96, kk, lane);
#pragma unroll
        for (int t = 0; t < 4; ++t) {
            v16h bf = frag_b(Wh, t*16, 96, kk, lane);
            acc[t] = __builtin_amdgcn_wmma_f32_16x16x32_f16(
                false, a, false, bf, (short)0, acc[t], false, false);
        }
    }

#pragma unroll
    for (int t = 0; t < 4; ++t) {
        int col = t*16 + lrow;
        float bia = b0[col];
        float s = 0.f, q = 0.f;
#pragma unroll
        for (int v = 0; v < 8; ++v) {
            float x = acc[t][v] + bia;
            s += x; q += x*x;
            int grow = row0 + wave*16 + lhalf*8 + v;
            h0[(size_t)grow*64 + col] = (half_t)x;
        }
        s += __shfl_xor(s, 16, 32);
        q += __shfl_xor(q, 16, 32);
        if (lhalf == 0) { sS[wave][col] = s; sQ[wave][col] = q; }
    }
    __syncthreads();
    if (tid < 64) {
        float s = 0.f, q = 0.f;
        for (int w = 0; w < 8; ++w) { s += sS[w][tid]; q += sQ[w][tid]; }
        atomicAdd(&sum0[tid], s);
        atomicAdd(&sq0[tid], q);
    }
}

// =====================================================================
// K4/K5: mid layer = BN+ReLU(prev stats) fused into A-load -> WMMA GEMM
// (K=64, CO=64 or 128) + bias, emit h_out (f16) + stats partials.
// =====================================================================
template <int CO>
__global__ __launch_bounds__(256) void pn2_gemm_mid(
    const half_t* __restrict__ hin,
    const float* __restrict__ sumP, const float* __restrict__ sqP,
    const float* __restrict__ gP, const float* __restrict__ betaP,
    const float* __restrict__ W, const float* __restrict__ bb,
    half_t* __restrict__ hout, float* __restrict__ sumO, float* __restrict__ sqO)
{
    __shared__ half_t Ah[128*64];
    __shared__ half_t Wh[CO*64];
    __shared__ float  sc[64], sh[64];
    __shared__ float  sS[8][CO];
    __shared__ float  sQ[8][CO];

    const int tid = threadIdx.x;
    if (tid < 64) {                       // finalize BN of previous layer
        float m   = sumP[tid] * (1.f / CNT_);
        float var = sqP[tid] * (1.f / CNT_) - m*m;
        float inv = rsqrtf(var + EPS_);
        float scl = gP[tid] * inv;
        sc[tid] = scl;
        sh[tid] = betaP[tid] - m*scl;
    }
    for (int i = tid; i < CO*64; i += 256) Wh[i] = (half_t)W[i];
    __syncthreads();

    const size_t base = (size_t)blockIdx.x * 128 * 64;
    for (int i = tid; i < 128*64; i += 256) {           // BN + ReLU fused
        int c = i & 63;
        float y = (float)hin[base + i] * sc[c] + sh[c];
        Ah[i] = (half_t)(y > 0.f ? y : 0.f);
    }
    __syncthreads();

    const int wave = tid >> 5, lane = tid & 31;
    const int lrow = lane & 15, lhalf = lane >> 4;
    const int row0 = blockIdx.x * 128;
    constexpr int NT = CO / 16;

    v8f acc[NT];
#pragma unroll
    for (int t = 0; t < NT; ++t)
#pragma unroll
        for (int j = 0; j < 8; ++j) acc[t][j] = 0.f;

#pragma unroll
    for (int kk = 0; kk < 64; kk += 32) {
        v16h a = frag_a(Ah, wave*16, 64, kk, lane);
#pragma unroll
        for (int t = 0; t < NT; ++t) {
            v16h bf = frag_b(Wh, t*16, 64, kk, lane);
            acc[t] = __builtin_amdgcn_wmma_f32_16x16x32_f16(
                false, a, false, bf, (short)0, acc[t], false, false);
        }
    }

#pragma unroll
    for (int t = 0; t < NT; ++t) {
        int col = t*16 + lrow;
        float bia = bb[col];
        float s = 0.f, q = 0.f;
#pragma unroll
        for (int v = 0; v < 8; ++v) {
            float x = acc[t][v] + bia;
            s += x; q += x*x;
            int grow = row0 + wave*16 + lhalf*8 + v;
            hout[(size_t)grow*CO + col] = (half_t)x;
        }
        s += __shfl_xor(s, 16, 32);
        q += __shfl_xor(q, 16, 32);
        if (lhalf == 0) { sS[wave][col] = s; sQ[wave][col] = q; }
    }
    __syncthreads();
    if (tid < CO) {
        float s = 0.f, q = 0.f;
        for (int w = 0; w < 8; ++w) { s += sS[w][tid]; q += sQ[w][tid]; }
        atomicAdd(&sumO[tid], s);
        atomicAdd(&sqO[tid], q);
    }
}

// =====================================================================
// K6: BN+ReLU on h2, max over NS samples, transpose to (B,128,NP)
// =====================================================================
__global__ __launch_bounds__(128) void pn2_final(
    const half_t* __restrict__ h2,
    const float* __restrict__ sum2, const float* __restrict__ sq2,
    const float* __restrict__ g2, const float* __restrict__ beta2,
    float* __restrict__ outF)
{
    const int c  = threadIdx.x;     // channel 0..127
    const int bp = blockIdx.x;      // center 0..B*NP-1
    float m   = sum2[c] * (1.f / CNT_);
    float var = sq2[c] * (1.f / CNT_) - m*m;
    float inv = rsqrtf(var + EPS_);
    float scl = g2[c] * inv;
    float shf = beta2[c] - m*scl;

    const half_t* p = h2 + (size_t)bp * NS_ * 128 + c;
    float best = -1e30f;
#pragma unroll 4
    for (int s = 0; s < NS_; ++s) {
        float y = (float)p[(size_t)s*128] * scl + shf;
        y = y > 0.f ? y : 0.f;
        best = best > y ? best : y;
    }
    int b = bp / NP_, pq = bp % NP_;
    outF[((size_t)b*128 + c) * NP_ + pq] = best;
}

// =====================================================================
extern "C" void kernel_launch(void* const* d_in, const int* in_sizes, int n_in,
                              void* d_out, int out_size, void* d_ws, size_t ws_size,
                              hipStream_t stream)
{
    (void)in_sizes; (void)n_in; (void)out_size; (void)ws_size;
    const float* xyz  = (const float*)d_in[0];
    const float* feat = (const float*)d_in[1];
    const float* W0 = (const float*)d_in[2];
    const float* b0 = (const float*)d_in[3];
    const float* g0 = (const float*)d_in[4];
    const float* be0= (const float*)d_in[5];
    const float* W1 = (const float*)d_in[6];
    const float* b1 = (const float*)d_in[7];
    const float* g1 = (const float*)d_in[8];
    const float* be1= (const float*)d_in[9];
    const float* W2 = (const float*)d_in[10];
    const float* b2 = (const float*)d_in[11];
    const float* g2 = (const float*)d_in[12];
    const float* be2= (const float*)d_in[13];

    float* outNewXyz = (float*)d_out;                       // (B,NP,3)
    float* outFeat   = (float*)d_out + (size_t)B_*NP_*3;    // (B,128,NP)

    char* ws = (char*)d_ws;
    int* fpsIdx  = (int*)ws;   ws += (size_t)B_*NP_*sizeof(int);
    int* ballIdx = (int*)ws;   ws += (size_t)B_*NP_*NS_*sizeof(int);
    float* stats = (float*)ws; ws += 512*sizeof(float);
    float *sum0 = stats,       *sq0 = stats + 64;
    float *sum1 = stats + 128, *sq1 = stats + 192;
    float *sum2 = stats + 256, *sq2 = stats + 384;
    half_t* h0 = (half_t*)ws;  ws += (size_t)CNT_*64*sizeof(half_t);
    half_t* h1 = (half_t*)ws;  ws += (size_t)CNT_*64*sizeof(half_t);
    half_t* h2 = (half_t*)ws;

    pn2_zero_stats<<<1, 512, 0, stream>>>(stats);
    pn2_fps<<<B_, 1024, 0, stream>>>(xyz, fpsIdx, outNewXyz);
    pn2_ball<<<(B_*NP_)/8, 256, 0, stream>>>(xyz, fpsIdx, ballIdx);
    pn2_gemm0<<<CNT_/128, 256, 0, stream>>>(xyz, feat, outNewXyz, ballIdx,
                                            W0, b0, h0, sum0, sq0);
    pn2_gemm_mid<64><<<CNT_/128, 256, 0, stream>>>(h0, sum0, sq0, g0, be0,
                                                   W1, b1, h1, sum1, sq1);
    pn2_gemm_mid<128><<<CNT_/128, 256, 0, stream>>>(h1, sum1, sq1, g1, be1,
                                                    W2, b2, h2, sum2, sq2);
    pn2_final<<<B_*NP_, 128, 0, stream>>>(h2, sum2, sq2, g2, be2, outFeat);
}
